// GAT_420906795145
// MI455X (gfx1250) — matrix operand order repaired and checked by reading, not compiled, not run on hardware
//
#include <hip/hip_runtime.h>

// ---------------------------------------------------------------------------
// GATv2 (2 layers, 8 heads, head-mean) for MI455X / gfx1250, wave32 + WMMA.
// ---------------------------------------------------------------------------

typedef __attribute__((ext_vector_type(16))) _Float16 v16h;
typedef __attribute__((ext_vector_type(8)))  _Float16 v8h;
typedef __attribute__((ext_vector_type(8)))  float    v8f;

#define N_NODES 50000
#define N_EDGES 400000
#define F_IN    256
#define HID     64
#define N_CLS   32
#define HEADS   8
#define E_TOT   (N_EDGES + N_NODES)   // self loops appended
#define NEG_SLOPE 0.2f

// ---------------------------------------------------------------------------
// WMMA GEMM: C[M,N] = A[M,K] * B[K,N], A/B fp32 row-major -> f16 in LDS,
// f32 accumulation via v_wmma_f32_16x16x32_f16. Block tile 128x64, K-step 32.
// 256 threads = 8 waves; wave w owns rows [16w,16w+16), 4 WMMA tiles wide.
// B fragments are preloaded so the 4 WMMAs issue back-to-back after a single
// DS wait instead of interleaving ds_load/wait/wmma.
// ---------------------------------------------------------------------------
#define TM 128
#define TN 64
#define TK 32
#define LDA (TK + 8)   // 40 elems = 80B row stride (keeps 16B alignment)

__global__ __launch_bounds__(256)
void gemm_f16wmma(const float* __restrict__ A, const float* __restrict__ B,
                  float* __restrict__ C, int M, int N, int K)
{
  __shared__ _Float16 As[TM][LDA];
  __shared__ _Float16 Bs[TN][LDA];   // transposed: Bs[n][k]

  const int tid  = threadIdx.x;
  const int wave = tid >> 5;
  const int lane = tid & 31;
  const int hlf  = lane >> 4;   // lane half (ISA A/B/C fragment layouts)
  const int ln   = lane & 15;

  const int m0 = blockIdx.y * TM;
  const int n0 = blockIdx.x * TN;

  v8f acc[4] = {};

  for (int k0 = 0; k0 < K; k0 += TK) {
    // ---- stage A tile: 128 x 32, each thread converts 16 elems ----
    {
      const int r  = tid >> 1;
      const int cb = (tid & 1) << 4;
      int gr = m0 + r; if (gr >= M) gr = M - 1;   // clamp; stores are guarded
      const float* src = A + (size_t)gr * K + k0 + cb;
      #pragma unroll
      for (int i = 0; i < 16; i += 4) {
        float4 v = *(const float4*)(src + i);
        As[r][cb + i + 0] = (_Float16)v.x;
        As[r][cb + i + 1] = (_Float16)v.y;
        As[r][cb + i + 2] = (_Float16)v.z;
        As[r][cb + i + 3] = (_Float16)v.w;
      }
      if (k0 + TK < K) __builtin_prefetch(src + TK, 0, 1);  // global_prefetch_b8
    }
    // ---- stage B tile transposed: Bs[n][k] = B[k0+k][n0+n] ----
    {
      const int k  = tid >> 3;            // 0..31
      const int nb = (tid & 7) << 3;      // 0..56
      const float* src = B + (size_t)(k0 + k) * N + n0;
      #pragma unroll
      for (int i = 0; i < 8; ++i) {
        const int n = nb + i;
        float v = (n0 + n < N) ? src[n] : 0.0f;
        Bs[n][k] = (_Float16)v;
      }
    }
    __syncthreads();

    // A fragment (16x32): lane half h holds K = h*8+j (j<8), 16+h*8+j (j>=8)
    const int am = (wave << 4) + ln;
    v8h alo = *(const v8h*)&As[am][hlf << 3];
    v8h ahi = *(const v8h*)&As[am][16 + (hlf << 3)];
    v16h a;
    #pragma unroll
    for (int i = 0; i < 8; ++i) { a[i] = alo[i]; a[i + 8] = ahi[i]; }

    // Preload all 4 B fragments (32x16 each: lane = col n, half h -> K=h*8+j (j<8), 16+h*8+j (j>=8))
    v16h bf[4];
    #pragma unroll
    for (int t = 0; t < 4; ++t) {
      v8h blo = *(const v8h*)&Bs[(t << 4) + ln][hlf << 3];
      v8h bhi = *(const v8h*)&Bs[(t << 4) + ln][16 + (hlf << 3)];
      #pragma unroll
      for (int i = 0; i < 8; ++i) { bf[t][i] = blo[i]; bf[t][i + 8] = bhi[i]; }
    }

    // Back-to-back WMMA issue
    #pragma unroll
    for (int t = 0; t < 4; ++t) {
      acc[t] = __builtin_amdgcn_wmma_f32_16x16x32_f16(
          false, a, false, bf[t], (short)0, acc[t], false, false);
    }
    __syncthreads();
  }

  // C fragment: c[v] = C[half*8 + v][lane%16]
  #pragma unroll
  for (int t = 0; t < 4; ++t) {
    const int col = n0 + (t << 4) + ln;
    if (col < N) {
      #pragma unroll
      for (int v = 0; v < 8; ++v) {
        const int row = m0 + (wave << 4) + (hlf << 3) + v;
        if (row < M) C[(size_t)row * N + col] = acc[t][v];
      }
    }
  }
}

// ---------------------------------------------------------------------------
// Edge-phase kernels (memory/atomic bound)
// ---------------------------------------------------------------------------
__device__ __forceinline__ unsigned enc_ord(float f) {
  unsigned u = __float_as_uint(f);
  return (u & 0x80000000u) ? ~u : (u | 0x80000000u);   // order-preserving map
}
__device__ __forceinline__ float dec_ord(unsigned u) {
  return __uint_as_float((u & 0x80000000u) ? (u & 0x7FFFFFFFu) : ~u);
}
__device__ __forceinline__ void edge_ep(const int* __restrict__ ei, int e,
                                        int& s, int& d) {
  if (e < N_EDGES) { s = ei[e]; d = ei[N_EDGES + e]; }
  else             { s = e - N_EDGES; d = s; }         // self loop
}

template<int C>
__global__ __launch_bounds__(256)
void edge_logits_kernel(const float* __restrict__ xl, const float* __restrict__ xr,
                        const float* __restrict__ att, const int* __restrict__ ei,
                        float* __restrict__ logits, unsigned* __restrict__ menc)
{
  const int idx = blockIdx.x * blockDim.x + threadIdx.x;
  if (idx >= E_TOT * HEADS) return;
  const int e = idx >> 3, h = idx & 7;
  int s, d; edge_ep(ei, e, s, d);
  const int D = HEADS * C;
  const float4* pl = (const float4*)(xl + (size_t)s * D + h * C);
  const float4* pr = (const float4*)(xr + (size_t)d * D + h * C);
  const float4* pa = (const float4*)(att + h * C);
  float acc = 0.0f;
  #pragma unroll
  for (int c = 0; c < C / 4; ++c) {
    float4 l = pl[c], r = pr[c], a = pa[c];
    float v;
    v = l.x + r.x; acc += a.x * (v > 0.f ? v : NEG_SLOPE * v);
    v = l.y + r.y; acc += a.y * (v > 0.f ? v : NEG_SLOPE * v);
    v = l.z + r.z; acc += a.z * (v > 0.f ? v : NEG_SLOPE * v);
    v = l.w + r.w; acc += a.w * (v > 0.f ? v : NEG_SLOPE * v);
  }
  logits[idx] = acc;
  atomicMax(&menc[d * HEADS + h], enc_ord(acc));       // segment max
}

__global__ __launch_bounds__(256)
void edge_exp_kernel(const int* __restrict__ ei, float* __restrict__ logits,
                     const unsigned* __restrict__ menc, float* __restrict__ denom)
{
  const int idx = blockIdx.x * blockDim.x + threadIdx.x;
  if (idx >= E_TOT * HEADS) return;
  const int e = idx >> 3, h = idx & 7;
  int s, d; edge_ep(ei, e, s, d);
  const float m = dec_ord(menc[d * HEADS + h]);
  const float p = __expf(logits[idx] - m);
  logits[idx] = p;                                     // overwrite with p
  atomicAdd(&denom[d * HEADS + h], p);                 // segment sum
}

template<int C>
__global__ __launch_bounds__(256)
void edge_accum_kernel(const float* __restrict__ xl, const int* __restrict__ ei,
                       const float* __restrict__ p, const float* __restrict__ denom,
                       float* __restrict__ out)
{
  const int idx = blockIdx.x * blockDim.x + threadIdx.x;
  if (idx >= E_TOT * HEADS) return;
  const int e = idx >> 3, h = idx & 7;
  int s, d; edge_ep(ei, e, s, d);
  const float alpha = p[idx] / denom[d * HEADS + h];
  const int D = HEADS * C;
  const float4* src = (const float4*)(xl + (size_t)s * D + h * C);
  float* o = out + (size_t)d * D + h * C;
  #pragma unroll
  for (int c = 0; c < C / 4; ++c) {
    float4 v = src[c];
    atomicAdd(o + 4 * c + 0, v.x * alpha);
    atomicAdd(o + 4 * c + 1, v.y * alpha);
    atomicAdd(o + 4 * c + 2, v.z * alpha);
    atomicAdd(o + 4 * c + 3, v.w * alpha);
  }
}

// ---------------------------------------------------------------------------
// Epilogues
// ---------------------------------------------------------------------------
__global__ __launch_bounds__(256)
void head_mean_relu_kernel(const float* __restrict__ acc, const float* __restrict__ bias,
                           float* __restrict__ out)
{
  const int idx = blockIdx.x * blockDim.x + threadIdx.x;
  if (idx >= N_NODES * HID) return;
  const int n = idx >> 6, c = idx & 63;
  float s = 0.0f;
  #pragma unroll
  for (int h = 0; h < HEADS; ++h) s += acc[(size_t)n * (HEADS * HID) + h * HID + c];
  s = s * (1.0f / HEADS) + bias[c];
  out[idx] = s > 0.f ? s : 0.f;
}

__global__ __launch_bounds__(256)
void final_combine_kernel(const float* __restrict__ acc, const float* __restrict__ b2,
                          const float* __restrict__ res, const float* __restrict__ blin,
                          float* __restrict__ out)
{
  const int idx = blockIdx.x * blockDim.x + threadIdx.x;
  if (idx >= N_NODES * N_CLS) return;
  const int n = idx >> 5, c = idx & 31;
  float s = 0.0f;
  #pragma unroll
  for (int h = 0; h < HEADS; ++h) s += acc[(size_t)n * (HEADS * N_CLS) + h * N_CLS + c];
  out[idx] = s * (1.0f / HEADS) + b2[c] + res[idx] + blin[c];
}

__global__ __launch_bounds__(256)
void fill_u32_kernel(unsigned* __restrict__ p, unsigned v, long long n)
{
  long long i = (long long)blockIdx.x * blockDim.x + threadIdx.x;
  if (i < n) p[i] = v;
}

// ---------------------------------------------------------------------------
// Host driver
// ---------------------------------------------------------------------------
extern "C" void kernel_launch(void* const* d_in, const int* in_sizes, int n_in,
                              void* d_out, int out_size, void* d_ws, size_t ws_size,
                              hipStream_t stream)
{
  (void)in_sizes; (void)n_in; (void)out_size; (void)ws_size;
  const float* x    = (const float*)d_in[0];
  const int*   ei   = (const int*)d_in[1];
  const float* Wl1  = (const float*)d_in[2];
  const float* Wr1  = (const float*)d_in[3];
  const float* att1 = (const float*)d_in[4];
  const float* b1   = (const float*)d_in[5];
  const float* Wl2  = (const float*)d_in[6];
  const float* Wr2  = (const float*)d_in[7];
  const float* att2 = (const float*)d_in[8];
  const float* b2   = (const float*)d_in[9];
  const float* Wlin = (const float*)d_in[10];
  const float* blin = (const float*)d_in[11];
  float* out = (float*)d_out;

  char* ws = (char*)d_ws;
  size_t off = 0;
  auto alloc = [&](size_t bytes) -> char* {
    char* p = ws + off;
    off += (bytes + 255) & ~(size_t)255;
    return p;
  };
  float*    XL  = (float*)alloc((size_t)N_NODES * 512 * 4);  // xl (L1:512, L2:256)
  float*    XR  = (float*)alloc((size_t)N_NODES * 512 * 4);  // xr
  float*    ACC = (float*)alloc((size_t)N_NODES * 512 * 4);  // scatter accum
  float*    LOG = (float*)alloc((size_t)E_TOT * HEADS * 4);  // logits -> p
  unsigned* MEN = (unsigned*)alloc((size_t)N_NODES * HEADS * 4);
  float*    DEN = (float*)alloc((size_t)N_NODES * HEADS * 4);
  float*    HB  = (float*)alloc((size_t)N_NODES * HID * 4);  // hidden h
  float*    RES = (float*)alloc((size_t)N_NODES * N_CLS * 4);

  auto gemm = [&](const float* A, const float* B, float* C, int M, int N, int K) {
    dim3 grid((N + TN - 1) / TN, (M + TM - 1) / TM);
    gemm_f16wmma<<<grid, dim3(256), 0, stream>>>(A, B, C, M, N, K);
  };
  auto fill = [&](void* p, unsigned v, size_t n) {
    fill_u32_kernel<<<(unsigned)((n + 255) / 256), 256, 0, stream>>>(
        (unsigned*)p, v, (long long)n);
  };

  const int eth = E_TOT * HEADS;
  const dim3 eg((eth + 255) / 256);

  // residual = x @ Wlin
  gemm(x, Wlin, RES, N_NODES, N_CLS, F_IN);

  // ---- layer 1 ----
  gemm(x, Wl1, XL, N_NODES, HEADS * HID, F_IN);
  gemm(x, Wr1, XR, N_NODES, HEADS * HID, F_IN);
  fill(MEN, 0u, (size_t)N_NODES * HEADS);
  fill(DEN, 0u, (size_t)N_NODES * HEADS);
  fill(ACC, 0u, (size_t)N_NODES * HEADS * HID);
  edge_logits_kernel<HID><<<eg, 256, 0, stream>>>(XL, XR, att1, ei, LOG, MEN);
  edge_exp_kernel<<<eg, 256, 0, stream>>>(ei, LOG, MEN, DEN);
  edge_accum_kernel<HID><<<eg, 256, 0, stream>>>(XL, ei, LOG, DEN, ACC);
  head_mean_relu_kernel<<<(N_NODES * HID + 255) / 256, 256, 0, stream>>>(ACC, b1, HB);

  // ---- layer 2 ----
  gemm(HB, Wl2, XL, N_NODES, HEADS * N_CLS, HID);
  gemm(HB, Wr2, XR, N_NODES, HEADS * N_CLS, HID);
  fill(MEN, 0u, (size_t)N_NODES * HEADS);
  fill(DEN, 0u, (size_t)N_NODES * HEADS);
  fill(ACC, 0u, (size_t)N_NODES * HEADS * N_CLS);
  edge_logits_kernel<N_CLS><<<eg, 256, 0, stream>>>(XL, XR, att2, ei, LOG, MEN);
  edge_exp_kernel<<<eg, 256, 0, stream>>>(ei, LOG, MEN, DEN);
  edge_accum_kernel<N_CLS><<<eg, 256, 0, stream>>>(XL, ei, LOG, DEN, ACC);
  final_combine_kernel<<<(N_NODES * N_CLS + 255) / 256, 256, 0, stream>>>(
      ACC, b2, RES, blin, out);
}
